// LSTMBaselineModel_70325794504851
// MI455X (gfx1250) — matrix operand order repaired and checked by reading, not compiled, not run on hardware
//
#include <hip/hip_runtime.h>
#include <math.h>

// ---------- types ----------
typedef __attribute__((ext_vector_type(16))) __bf16 v16bf;
typedef __attribute__((ext_vector_type(8)))  __bf16 v8bf;
typedef __attribute__((ext_vector_type(8)))  float  v8f;

#define SEQ    128
#define BATCH  256
#define INPUT  300
#define KXPAD  320              // x K padded to 10 k-tiles of 32
#define HID    1024
#define KTOT   (KXPAD + HID)    // 1344 = 42 k-tiles
#define NG     (4 * HID)        // 4096

// ---------- WMMA fragment loaders (layouts per CDNA5 ISA §7.12.2) ----------
// A: 16x32 bf16 (MxK). lane holds row M=lane%16; half=lane/16:
//   V0..3 = K[half*8 .. half*8+7], V4..7 = K[16+half*8 .. +7]  -> two b128 loads
__device__ __forceinline__ v16bf ldfragA(const __bf16* __restrict__ base, int stride,
                                         int m0, int k0) {
  const int lane = threadIdx.x & 31;
  const int half = lane >> 4;
  const __bf16* p = base + (size_t)(m0 + (lane & 15)) * stride + k0 + half * 8;
  v8bf lo = *(const v8bf*)p;          // K = half*8 .. +7
  v8bf hi = *(const v8bf*)(p + 16);   // K = 16+half*8 .. +7
  v16bf r;
#pragma unroll
  for (int i = 0; i < 8; ++i) { r[i] = lo[i]; r[i + 8] = hi[i]; }
  return r;
}

// B: 32x16 bf16 (KxN), B[k][n] = W[n][k] (W row-major [N][K]).
// lane holds col N=lane%16; half=lane/16: V0..3 = K[half*8 .. half*8+7], V4..7 = K[16+half*8 .. +7] -> two b128 loads
__device__ __forceinline__ v16bf ldfragB(const __bf16* __restrict__ Wrow, int stride,
                                         int n0, int k0) {
  const int lane = threadIdx.x & 31;
  const int half = lane >> 4;
  const __bf16* p = Wrow + (size_t)(n0 + (lane & 15)) * stride + k0 + half * 8;
  v8bf lo = *(const v8bf*)p;          // K = half*8 .. +7
  v8bf hi = *(const v8bf*)(p + 16);   // K = 16+half*8 .. +7
  v16bf r;
#pragma unroll
  for (int i = 0; i < 8; ++i) { r[i] = lo[i]; r[i + 8] = hi[i]; }
  return r;
}

__device__ __forceinline__ void mma4(const v16bf& a, const __bf16* __restrict__ Wc,
                                     int ncol, int k0,
                                     v8f& c0, v8f& c1, v8f& c2, v8f& c3) {
  v16bf b0 = ldfragB(Wc, KTOT, 0 * HID + ncol, k0);
  v16bf b1 = ldfragB(Wc, KTOT, 1 * HID + ncol, k0);
  v16bf b2 = ldfragB(Wc, KTOT, 2 * HID + ncol, k0);
  v16bf b3 = ldfragB(Wc, KTOT, 3 * HID + ncol, k0);
  c0 = __builtin_amdgcn_wmma_f32_16x16x32_bf16(false, a, false, b0, (short)0, c0, false, false);
  c1 = __builtin_amdgcn_wmma_f32_16x16x32_bf16(false, a, false, b1, (short)0, c1, false, false);
  c2 = __builtin_amdgcn_wmma_f32_16x16x32_bf16(false, a, false, b2, (short)0, c2, false, false);
  c3 = __builtin_amdgcn_wmma_f32_16x16x32_bf16(false, a, false, b3, (short)0, c3, false, false);
}

__device__ __forceinline__ float fsigmoid(float x) { return 1.f / (1.f + __expf(-x)); }
__device__ __forceinline__ float ftanh(float x) {
  float a = fabsf(x);
  float e = __expf(-2.f * a);
  float t = (1.f - e) / (1.f + e);
  return copysignf(t, x);
}

// ---------- prep kernels (bf16 conversion, run every launch: deterministic) ----------
__global__ void prep_wc(const float* __restrict__ Wih, const float* __restrict__ Whh,
                        __bf16* __restrict__ Wc) {
  size_t idx = (size_t)blockIdx.x * blockDim.x + threadIdx.x;
  if (idx >= (size_t)NG * KTOT) return;
  int n = (int)(idx / KTOT), k = (int)(idx % KTOT);
  float v;
  if (k < INPUT)      v = Wih[(size_t)n * INPUT + k];
  else if (k < KXPAD) v = 0.f;
  else                v = Whh[(size_t)n * HID + (k - KXPAD)];
  Wc[idx] = (__bf16)v;
}

__global__ void prep_xp(const float* __restrict__ x, __bf16* __restrict__ Xp) {
  size_t idx = (size_t)blockIdx.x * blockDim.x + threadIdx.x;
  if (idx >= (size_t)SEQ * BATCH * KXPAD) return;
  size_t row = idx / KXPAD;
  int k = (int)(idx % KXPAD);
  Xp[idx] = (__bf16)(k < INPUT ? x[row * INPUT + k] : 0.f);
}

__global__ void prep_w1(const float* __restrict__ W1, __bf16* __restrict__ W1b) {
  size_t idx = (size_t)blockIdx.x * blockDim.x + threadIdx.x;
  if (idx < (size_t)HID * HID) W1b[idx] = (__bf16)W1[idx];
}

__global__ void prep_bias(const float* __restrict__ bih, const float* __restrict__ bhh,
                          float* __restrict__ bias) {
  int i = blockIdx.x * blockDim.x + threadIdx.x;
  if (i < NG) bias[i] = bih[i] + bhh[i];
}

__global__ void init_state(float* __restrict__ C, unsigned* __restrict__ h0w) {
  int i = blockIdx.x * blockDim.x + threadIdx.x;
  if (i < BATCH * HID) C[i] = 0.f;
  if (i < BATCH * HID / 2) h0w[i] = 0u;   // zero bf16 h0 (2 per dword)
}

// ---------- fused LSTM step: gates GEMM (WMMA) + pointwise c/h update ----------
// grid 256 blocks x 128 threads (4 waves). wave-task: msub(16) x nsub(64).
__global__ __launch_bounds__(128) void lstm_step(
    const __bf16* __restrict__ Xp, const __bf16* __restrict__ Wc,
    const float* __restrict__ bias, const __bf16* __restrict__ hin,
    __bf16* __restrict__ hout, float* __restrict__ C, int t) {
  const int wave = threadIdx.x >> 5;
  const int lane = threadIdx.x & 31;
  const int task = blockIdx.x * 4 + wave;
  const int mbase = (task & 15) * 16;
  const int ncol  = (task >> 4) * 16;

  v8f a0 = {}, a1 = {}, a2 = {}, a3 = {};
  const __bf16* xb = Xp + (size_t)t * BATCH * KXPAD;

#pragma unroll
  for (int kt = 0; kt < KXPAD / 32; ++kt) {           // x-sourced K range
    v16bf a = ldfragA(xb, KXPAD, mbase, kt * 32);
    mma4(a, Wc, ncol, kt * 32, a0, a1, a2, a3);
  }
#pragma unroll 4
  for (int kt = KXPAD / 32; kt < KTOT / 32; ++kt) {   // h-sourced K range
    v16bf a = ldfragA(hin, HID, mbase, kt * 32 - KXPAD);
    mma4(a, Wc, ncol, kt * 32, a0, a1, a2, a3);
  }

  // C/D layout: lane holds col n=lane%16; rows mbase + half*8 + r in VGPR r
  const int half = lane >> 4;
  const int n = ncol + (lane & 15);
  const float bi = bias[n], bff = bias[HID + n], bg = bias[2 * HID + n], bo = bias[3 * HID + n];
#pragma unroll
  for (int r = 0; r < 8; ++r) {
    int m = mbase + half * 8 + r;
    float si = fsigmoid(a0[r] + bi);
    float sf = fsigmoid(a1[r] + bff);
    float tg = ftanh(a2[r] + bg);
    float so = fsigmoid(a3[r] + bo);
    size_t idx = (size_t)m * HID + n;
    float c = sf * C[idx] + si * tg;
    C[idx] = c;
    hout[idx] = (__bf16)(so * ftanh(c));
  }
}

// ---------- head: z = h @ W1^T + b1 (WMMA) ----------
__global__ __launch_bounds__(128) void head_gemm(
    const __bf16* __restrict__ H, const __bf16* __restrict__ W1b,
    const float* __restrict__ b1, float* __restrict__ Z) {
  const int wave = threadIdx.x >> 5;
  const int lane = threadIdx.x & 31;
  const int task = blockIdx.x * 4 + wave;
  const int mbase = (task & 15) * 16;
  const int ncol  = (task >> 4) * 16;
  v8f acc = {};
#pragma unroll 4
  for (int kt = 0; kt < HID / 32; ++kt) {
    v16bf a = ldfragA(H, HID, mbase, kt * 32);
    v16bf b = ldfragB(W1b, HID, ncol, kt * 32);
    acc = __builtin_amdgcn_wmma_f32_16x16x32_bf16(false, a, false, b, (short)0, acc, false, false);
  }
  const int half = lane >> 4;
  const int n = ncol + (lane & 15);
  const float bb = b1[n];
#pragma unroll
  for (int r = 0; r < 8; ++r)
    Z[(size_t)(mbase + half * 8 + r) * HID + n] = acc[r] + bb;
}

// ---------- batch-norm stats (train-mode batch stats) ----------
__global__ void bn_stats(const float* __restrict__ Z, const float* __restrict__ gamma,
                         const float* __restrict__ beta, float* __restrict__ scale,
                         float* __restrict__ shift) {
  int j = blockIdx.x * blockDim.x + threadIdx.x;
  if (j >= HID) return;
  float s = 0.f, s2 = 0.f;
  for (int b = 0; b < BATCH; ++b) {
    float v = Z[(size_t)b * HID + j];
    s += v; s2 += v * v;
  }
  float mean = s * (1.f / BATCH);
  float var  = s2 * (1.f / BATCH) - mean * mean;
  float sc = gamma[j] * rsqrtf(var + 1e-5f);
  scale[j] = sc;
  shift[j] = beta[j] - mean * sc;
}

// ---------- final: relu(BN(z)) @ W2^T + b2 -> 3*sigmoid ----------
__global__ __launch_bounds__(256) void head_final(
    const float* __restrict__ Z, const float* __restrict__ scale,
    const float* __restrict__ shift, const float* __restrict__ W2,
    const float* __restrict__ b2, float* __restrict__ out) {
  __shared__ float red[256];
  const int b = blockIdx.x, tid = threadIdx.x;
  float s = 0.f;
  for (int j = tid; j < HID; j += 256) {
    float v = Z[(size_t)b * HID + j] * scale[j] + shift[j];
    v = v > 0.f ? v : 0.f;
    s += v * W2[j];
  }
  red[tid] = s;
  __syncthreads();
  for (int o = 128; o > 0; o >>= 1) {
    if (tid < o) red[tid] += red[tid + o];
    __syncthreads();
  }
  if (tid == 0) out[b] = 3.f * fsigmoid(red[0] + b2[0]);
}

// ---------- launch ----------
extern "C" void kernel_launch(void* const* d_in, const int* in_sizes, int n_in,
                              void* d_out, int out_size, void* d_ws, size_t ws_size,
                              hipStream_t stream) {
  const float* x    = (const float*)d_in[0];
  const float* Wih  = (const float*)d_in[1];
  const float* Whh  = (const float*)d_in[2];
  const float* bih  = (const float*)d_in[3];
  const float* bhh  = (const float*)d_in[4];
  const float* W1   = (const float*)d_in[5];
  const float* b1   = (const float*)d_in[6];
  const float* gam  = (const float*)d_in[7];
  const float* bet  = (const float*)d_in[8];
  const float* W2   = (const float*)d_in[9];
  const float* b2   = (const float*)d_in[10];
  float* out = (float*)d_out;

  char* ws = (char*)d_ws;
  size_t o = 0;
  __bf16* Wc  = (__bf16*)(ws + o); o += (size_t)NG * KTOT * 2;            // 11.0 MB
  __bf16* Xp  = (__bf16*)(ws + o); o += (size_t)SEQ * BATCH * KXPAD * 2;  // 21.0 MB
  __bf16* W1b = (__bf16*)(ws + o); o += (size_t)HID * HID * 2;            //  2.1 MB
  float*  bias = (float*)(ws + o); o += (size_t)NG * 4;                   // 16 KB
  __bf16* h0  = (__bf16*)(ws + o); o += (size_t)BATCH * HID * 2;          // 0.5 MB
  __bf16* h1  = (__bf16*)(ws + o); o += (size_t)BATCH * HID * 2;          // 0.5 MB
  float*  C   = (float*)(ws + o);  o += (size_t)BATCH * HID * 4;          // 1.0 MB
  float*  Z   = (float*)(ws + o);  o += (size_t)BATCH * HID * 4;          // 1.0 MB
  float*  scale = (float*)(ws + o); o += (size_t)HID * 4;
  float*  shift = (float*)(ws + o); o += (size_t)HID * 4;

  // prep (re-run every call: deterministic, ws may be poisoned)
  {
    size_t n = (size_t)NG * KTOT;
    prep_wc<<<(unsigned)((n + 255) / 256), 256, 0, stream>>>(Wih, Whh, Wc);
  }
  {
    size_t n = (size_t)SEQ * BATCH * KXPAD;
    prep_xp<<<(unsigned)((n + 255) / 256), 256, 0, stream>>>(x, Xp);
  }
  prep_w1<<<(HID * HID + 255) / 256, 256, 0, stream>>>(W1, W1b);
  prep_bias<<<(NG + 255) / 256, 256, 0, stream>>>(bih, bhh, bias);
  init_state<<<(BATCH * HID + 255) / 256, 256, 0, stream>>>(C, (unsigned*)h0);

  // recurrence: 128 stream-ordered fused steps, double-buffered h (bf16)
  for (int t = 0; t < SEQ; ++t) {
    const __bf16* hin = (t & 1) ? h1 : h0;
    __bf16* hout      = (t & 1) ? h0 : h1;
    lstm_step<<<256, 128, 0, stream>>>(Xp, Wc, bias, hin, hout, C, t);
  }
  // after t=127 (odd), final h is in h0

  head_gemm<<<256, 128, 0, stream>>>(h0, W1b, b1, Z);
  bn_stats<<<(HID + 255) / 256, 256, 0, stream>>>(Z, gam, bet, scale, shift);
  head_final<<<BATCH, 256, 0, stream>>>(Z, scale, shift, W2, b2, out);
}
